// Hippocampus_1675037245790
// MI455X (gfx1250) — matrix operand order repaired and checked by reading, not compiled, not run on hardware
//
#include <hip/hip_runtime.h>
#include <math.h>

// Problem constants: B=8, T=1024, D=2048, H=16, DK=DV=64
// M = B*T = 8192, Kdim = 2048, N = H*DK + H*DV + H = 2064

typedef __attribute__((ext_vector_type(16))) __bf16 v16bf;
typedef __attribute__((ext_vector_type(8)))  __bf16 v8bf;
typedef __attribute__((ext_vector_type(8)))  float  v8f;
typedef __attribute__((ext_vector_type(4)))  unsigned int v4u;
typedef __attribute__((ext_vector_type(8)))  int v8i;
typedef __attribute__((ext_vector_type(4)))  int v4i;

// A-fragment (16x32 bf16): lane m (<16) holds K {0..7,16..23}; lane m+16 holds {8..15,24..31}.
// p = row_base + k0 + (hi ? 8 : 0); pulls [p..p+7] and [p+16..p+23].
__device__ __forceinline__ v16bf load_a_frag(const __bf16* p) {
  v8bf lo = *reinterpret_cast<const v8bf*>(p);
  v8bf hi = *reinterpret_cast<const v8bf*>(p + 16);
  v16bf f;
#pragma unroll
  for (int e = 0; e < 8; ++e) { f[e] = lo[e]; f[8 + e] = hi[e]; }
  return f;
}
// B-fragment (32x16 bf16), B[k,n] = srcRow_n[k]: lane n (<16) holds K {0..7,16..23}, lane n+16 holds {8..15,24..31}.
// p = row_base + k0 + (hi ? 8 : 0); pulls [p..p+7] and [p+16..p+23].
__device__ __forceinline__ v16bf load_b_frag(const __bf16* p) {
  v8bf lo = *reinterpret_cast<const v8bf*>(p);
  v8bf hi = *reinterpret_cast<const v8bf*>(p + 16);
  v16bf f;
#pragma unroll
  for (int e = 0; e < 8; ++e) { f[e] = lo[e]; f[8 + e] = hi[e]; }
  return f;
}
__device__ __forceinline__ v8f wmma_bf16(v16bf a, v16bf b, v8f c) {
  return __builtin_amdgcn_wmma_f32_16x16x32_bf16(false, a, false, b, (short)0, c, false, false);
}

// TDM: contiguous nbytes=8192 global -> LDS tile via Tensor Data Mover (D# per ISA 08 §8).
// 2-D tensor, data_size=2B, tensor_dim0 = tile_dim0 = 4096 elems, dims1 = 1, groups 2/3 zero.
__device__ __forceinline__ void tdm_load_8k(const void* gptr, void* lptr) {
  const unsigned int n = 4096;  // elements (2 bytes each)
  unsigned long long ga = (unsigned long long)(uintptr_t)gptr;
  unsigned int la = (unsigned int)(uintptr_t)lptr;  // generic addr low 32 bits == LDS byte offset
  v4u g0 = { 1u,                                       // count=1, no gather
             la,                                       // lds_addr
             (unsigned int)ga,                         // global_addr[31:0]
             (unsigned int)(ga >> 32) | (2u << 30) };  // global_addr[56:32] | type=2
  v8i g1 = { (int)0x00010000u,                         // wg_mask=0, data_size=1 (2B)
             (int)((n & 0xFFFFu) << 16),               // tensor_dim0[15:0] @ bits63:48
             (int)(((n >> 16) & 0xFFFFu) | (1u << 16)),// tensor_dim0[31:16] | tensor_dim1[15:0]=1
             (int)((n & 0xFFFFu) << 16),               // tensor_dim1[31:16]=0 | tile_dim0=n
             (int)1,                                   // tile_dim1=1, tile_dim2=0
             (int)n,                                   // tensor_dim0_stride[31:0]
             0, 0 };
  v4i z4 = {0, 0, 0, 0};
#if __clang_major__ >= 23
  v8i z8 = {0, 0, 0, 0, 0, 0, 0, 0};
  __builtin_amdgcn_tensor_load_to_lds(g0, g1, z4, z4, z8, 0);
#else
  __builtin_amdgcn_tensor_load_to_lds(g0, g1, z4, z4, 0);
#endif
}

// ---------------- Kernel 0: f32 -> bf16 convert ----------------
__global__ void f32_to_bf16(const float* __restrict__ s, __bf16* __restrict__ d, int n) {
  int i = blockIdx.x * blockDim.x + threadIdx.x;
  if (i < n) d[i] = (__bf16)s[i];
}

// ---------------- Kernel 1: projection GEMM + fused epilogues ----------------
// C[M=8192, N=2064] = content_bf16 @ Wcat_bf16^T.
// grid = (512, 9); block = 128 (4 waves). Each WAVE owns a 16(M) x 64(N) strip
// (4 accumulators, A-fragment reused 4x). Strip base n0 is head-aligned:
//   n0 <  1024 : k head h=n0/64 -> l2-normalize rows via shfl_xor butterflies, store bf16
//   n0 <  2048 : v              -> store f32
//   n0 == 2048 : beta (16 cols; tile 0 only) -> sigmoid(x + b_b[h])
__global__ __launch_bounds__(128) void proj_gemm(
    const __bf16* __restrict__ cbf, const __bf16* __restrict__ wbf,
    const float* __restrict__ b_b,
    __bf16* __restrict__ k_out, float* __restrict__ v_out, float* __restrict__ beta_out)
{
  const int tid  = threadIdx.x;
  const int wid  = tid >> 5;
  const int lane = tid & 31;
  const int lr   = lane & 15;
  const bool hih = lane >= 16;
  const int m0 = blockIdx.x * 16;
  const int n0raw = (blockIdx.y * 4 + wid) * 64;
  const int n0 = n0raw > 2048 ? 2048 : n0raw;

  const __bf16* arow = cbf + (size_t)(m0 + lr) * 2048 + (hih ? 8 : 0);
  const __bf16* brow[4];
#pragma unroll
  for (int t = 0; t < 4; ++t) {
    int r = n0 + t * 16 + lr;
    if (r > 2063) r = 2063;  // clamp OOB rows of the beta group (no store for them)
    brow[t] = wbf + (size_t)r * 2048 + (hih ? 8 : 0);
  }

  v8f acc[4] = {{}, {}, {}, {}};
#pragma unroll 2
  for (int kk = 0; kk < 64; ++kk) {
    v16bf a = load_a_frag(arow + kk * 32);
#pragma unroll
    for (int t = 0; t < 4; ++t) {
      v16bf b = load_b_frag(brow[t] + kk * 32);
      acc[t] = wmma_bf16(a, b, acc[t]);
    }
  }

  if (n0 < 1024) {
    // k head: per-row sum of squares across 64 cols = 4 tiles * 16 lanes (half-wave butterflies)
    const int h = n0 >> 6;
    float scale[8];
#pragma unroll
    for (int vv = 0; vv < 8; ++vv) {
      float s = acc[0][vv] * acc[0][vv] + acc[1][vv] * acc[1][vv] +
                acc[2][vv] * acc[2][vv] + acc[3][vv] * acc[3][vv];
      s += __shfl_xor(s, 1, 32);
      s += __shfl_xor(s, 2, 32);
      s += __shfl_xor(s, 4, 32);
      s += __shfl_xor(s, 8, 32);
      scale[vv] = 1.0f / fmaxf(sqrtf(s), 1e-12f);
    }
#pragma unroll
    for (int vv = 0; vv < 8; ++vv) {
      int mg = m0 + vv + (hih ? 8 : 0);
      int b_ = mg >> 10, t_ = mg & 1023;
      size_t rowb = (((size_t)(b_ * 16 + h)) * 1024 + t_) * 64;
#pragma unroll
      for (int t = 0; t < 4; ++t)
        k_out[rowb + t * 16 + lr] = (__bf16)(acc[t][vv] * scale[vv]);
    }
  } else if (n0 < 2048) {
    const int h = (n0 - 1024) >> 6;
#pragma unroll
    for (int vv = 0; vv < 8; ++vv) {
      int mg = m0 + vv + (hih ? 8 : 0);
      int b_ = mg >> 10, t_ = mg & 1023;
      size_t rowb = (((size_t)(b_ * 16 + h)) * 1024 + t_) * 64;
#pragma unroll
      for (int t = 0; t < 4; ++t)
        v_out[rowb + t * 16 + lr] = acc[t][vv];
    }
  } else if (n0raw == 2048) {
    const int h = lr;
    float bb = b_b[h];
#pragma unroll
    for (int vv = 0; vv < 8; ++vv) {
      int mg = m0 + vv + (hih ? 8 : 0);
      int b_ = mg >> 10, t_ = mg & 1023;
      float x = acc[0][vv] + bb;
      beta_out[((size_t)(b_ * 16 + h)) * 1024 + t_] = 1.0f / (1.0f + expf(-x));
    }
  }
}

// ---------------- Kernel 2: chunkwise delta-rule recurrence ----------------
// One block per (b,h); 512 threads = 16 waves; 16 sequential chunks of C=64.
// Kc chunk (8KB contiguous) fetched by the Tensor Data Mover (wave 0 issues,
// waits on TENSORcnt, block barrier publishes LDS).
__global__ __launch_bounds__(512) void delta_chunk(
    const float* __restrict__ W_old, const __bf16* __restrict__ k_bf,
    const float* __restrict__ v_f, const float* __restrict__ beta_f,
    float* __restrict__ W_out)
{
  __shared__ float  Wl[64 * 64];
  __shared__ __bf16 Wb[64 * 64] __attribute__((aligned(16)));
  __shared__ __bf16 Kc[64 * 64] __attribute__((aligned(16)));
  __shared__ float  Al[64 * 64];
  __shared__ float  RV[64 * 64];
  __shared__ float  betaL[64];

  const int tid  = threadIdx.x;
  const int wid  = tid >> 5;
  const int lane = tid & 31;
  const int lr   = lane & 15;
  const bool hih = lane >= 16;
  const int ti = wid >> 2, tj = wid & 3;    // 4x4 tile grid over 64x64
  const int bh = blockIdx.x;
  const size_t base_td = (size_t)bh * 1024 * 64;

  for (int i = tid; i < 4096; i += 512) {
    float w = W_old[(size_t)bh * 4096 + i];
    Wl[i] = w; Wb[i] = (__bf16)w;
  }
  __syncthreads();

  for (int c = 0; c < 16; ++c) {
    const int t0 = c * 64;
    if (wid == 0) {
      tdm_load_8k(k_bf + base_td + (size_t)t0 * 64, Kc);
      __builtin_amdgcn_s_wait_tensorcnt(0);
    }
    if (tid < 64) betaL[tid] = beta_f[(size_t)bh * 1024 + t0 + tid];
    __syncthreads();

    // Step A: R[t,v] = V[t,v] - sum_d Kc[t,d] * W[v,d]
    {
      v8f acc = {};
#pragma unroll
      for (int k0 = 0; k0 < 64; k0 += 32) {
        v16bf a = load_a_frag(&Kc[(16 * ti + lr) * 64 + k0 + (hih ? 8 : 0)]);
        v16bf b = load_b_frag(&Wb[(16 * tj + lr) * 64 + k0 + (hih ? 8 : 0)]);
        acc = wmma_bf16(a, b, acc);
      }
#pragma unroll
      for (int vv = 0; vv < 8; ++vv) {
        int t = 16 * ti + vv + (hih ? 8 : 0);
        int v = 16 * tj + lr;
        RV[t * 64 + v] = v_f[base_td + (size_t)(t0 + t) * 64 + v] - acc[vv];
      }
    }
    // Step B: Al[i,j] = beta_j * (k_i . k_j)
    {
      v8f acc = {};
#pragma unroll
      for (int k0 = 0; k0 < 64; k0 += 32) {
        v16bf a = load_a_frag(&Kc[(16 * ti + lr) * 64 + k0 + (hih ? 8 : 0)]);
        v16bf b = load_b_frag(&Kc[(16 * tj + lr) * 64 + k0 + (hih ? 8 : 0)]);
        acc = wmma_bf16(a, b, acc);
      }
#pragma unroll
      for (int vv = 0; vv < 8; ++vv) {
        int i = 16 * ti + vv + (hih ? 8 : 0);
        int j = 16 * tj + lr;
        Al[i * 64 + j] = acc[vv] * betaL[j];
      }
    }
    __syncthreads();

    // Forward substitution: Vp[i] = R[i] - sum_{j<i} Al[i,j] * Vp[j]   (in-place in RV, f32)
    for (int j = 0; j < 63; ++j) {
      int tot = (63 - j) * 64;
      for (int idx = tid; idx < tot; idx += 512) {
        int i = j + 1 + (idx >> 6);
        int v = idx & 63;
        RV[i * 64 + v] -= Al[i * 64 + j] * RV[j * 64 + v];
      }
      __syncthreads();
    }

    // Step E: W[v,d] += sum_t (beta_t * Vp[t,v]) * Kc[t,d]
    {
      v8f acc = {};
#pragma unroll
      for (int k0 = 0; k0 < 64; k0 += 32) {
        int vcol = 16 * ti + lr;
        int kbA = k0 + (hih ? 8 : 0);
        v16bf af;
#pragma unroll
        for (int e = 0; e < 8; ++e) {
          int t1 = kbA + e;
          af[e] = (__bf16)(betaL[t1] * RV[t1 * 64 + vcol]);
          int t2 = kbA + 16 + e;
          af[8 + e] = (__bf16)(betaL[t2] * RV[t2 * 64 + vcol]);
        }
        int dcol = 16 * tj + lr;
        int kbB = k0 + (hih ? 8 : 0);
        v16bf bf2;
#pragma unroll
        for (int e = 0; e < 16; ++e) bf2[e] = Kc[(kbB + e + (e >= 8 ? 8 : 0)) * 64 + dcol];
        acc = wmma_bf16(af, bf2, acc);
      }
#pragma unroll
      for (int vv = 0; vv < 8; ++vv) {
        int vrow = 16 * ti + vv + (hih ? 8 : 0);
        int d = 16 * tj + lr;
        int idx = vrow * 64 + d;
        float nw = Wl[idx] + acc[vv];
        Wl[idx] = nw;
        Wb[idx] = (__bf16)nw;
      }
    }
    __syncthreads();
  }

  for (int i = tid; i < 4096; i += 512)
    W_out[(size_t)bh * 4096 + i] = Wl[i];
}

// ---------------- launcher ----------------
extern "C" void kernel_launch(void* const* d_in, const int* in_sizes, int n_in,
                              void* d_out, int out_size, void* d_ws, size_t ws_size,
                              hipStream_t stream) {
  const float* W_old   = (const float*)d_in[0];
  const float* content = (const float*)d_in[1];
  const float* k_w     = (const float*)d_in[2];
  const float* v_w     = (const float*)d_in[3];
  const float* b_w     = (const float*)d_in[4];
  const float* b_b     = (const float*)d_in[5];

  // workspace layout (bytes):
  //   content_bf16 : 8192*2048*2  = 33,554,432
  //   Wcat_bf16    : 2064*2048*2  =  8,454,144
  //   k_bf16       : 128*1024*64*2 = 16,777,216
  //   v_f32        : 128*1024*64*4 = 33,554,432
  //   beta_f32     : 128*1024*4    =     524,288      total ~93 MB
  char* ws = (char*)d_ws;
  __bf16* cbf    = (__bf16*)ws;
  __bf16* wbf    = (__bf16*)(ws + 33554432);
  __bf16* kbf    = (__bf16*)(ws + 33554432 + 8454144);
  float*  vws    = (float*)(ws + 33554432 + 8454144 + 16777216);
  float*  betaws = (float*)(ws + 33554432 + 8454144 + 16777216 + 33554432);

  const int nC = 8192 * 2048;
  f32_to_bf16<<<(nC + 255) / 256, 256, 0, stream>>>(content, cbf, nC);
  const int nW = 1024 * 2048;
  f32_to_bf16<<<(nW + 255) / 256, 256, 0, stream>>>(k_w, wbf, nW);
  f32_to_bf16<<<(nW + 255) / 256, 256, 0, stream>>>(v_w, wbf + nW, nW);
  const int nB = 16 * 2048;
  f32_to_bf16<<<(nB + 255) / 256, 256, 0, stream>>>(b_w, wbf + 2 * nW, nB);

  proj_gemm<<<dim3(512, 9), 128, 0, stream>>>(cbf, wbf, b_b, kbf, vws, betaws);
  delta_chunk<<<128, 512, 0, stream>>>(W_old, kbf, vws, betaws, (float*)d_out);
}
